// FrameNetLigand01_44117904065152
// MI455X (gfx1250) — compile-verified, matrix operand order checked by reading, not run on hardware
//
#include <hip/hip_runtime.h>
#include <math.h>

// ---------------------------------------------------------------------------
// FrameNet-style ligand GNN forward for gfx1250 (MI455X).
// Dense layers run through a WMMA (v_wmma_f32_16x16x32_f16) tiled GEMM with
// an LDS-staged B panel; hot activation streams are stored f16.
// A-precision is a template parameter -> branch-free inner loop.
// ---------------------------------------------------------------------------

#define HDIM     128
#define RBF_N    96
#define NLAYERS  4
#define NNODES   8000
#define NEDGES   128000
#define DIN      480      // 3*H + R
#define DHALF    240      // din // 2

#define INV_SQRT3 0.57735026918962576451f
#define INV_SQRTH 0.08838834764831845f   // 1/sqrt(128)
#define INV_SQRT2 0.70710678118654752440f
#define PI_F      3.14159265358979323846f

typedef __attribute__((ext_vector_type(16))) _Float16 v16h;
typedef __attribute__((ext_vector_type(8)))  _Float16 v8h;
typedef __attribute__((ext_vector_type(8)))  float    v8f;

__device__ __forceinline__ float silu_f(float x) {
    return x * (1.0f / (1.0f + __expf(-x)));
}

// ---------------------------------------------------------------------------
// Weight packing: rearrange a KxN f32 weight into the B-fragment layout of
// V_WMMA_F32_16X16X32_F16 (ISA 7.12.2).  Fragment = 32 K x 16 N.  Lane L
// holds column N = n_tile*16 + (L%16); lane group g = L/16 holds K in
// {8g..8g+7} U {16+8g..16+8g+7}.  Each lane's 16 halves are contiguous (32B).
// fragIdx = n_tile * (Kp/32) + k_tile  =>  a block's 4-n-tile panel is one
// contiguous [4*kt*512] run, ideal for linear LDS staging.
// ---------------------------------------------------------------------------
__global__ void pack_w_kernel(const float* __restrict__ W, _Float16* __restrict__ P,
                              int K, int Nw, int Kp, int Np) {
    long long idx = (long long)blockIdx.x * blockDim.x + threadIdx.x;
    long long total = (long long)(Kp / 32) * (Np / 16) * 512;
    if (idx >= total) return;
    int i    = (int)(idx & 15);
    int lane = (int)((idx >> 4) & 31);
    long long f = idx >> 9;
    int kt     = Kp / 32;
    int k_tile = (int)(f % kt);
    int n_tile = (int)(f / kt);
    int n = n_tile * 16 + (lane & 15);
    int g = lane >> 4;
    int v = i >> 1, lo = i & 1;
    int k = k_tile * 32 + ((v < 4) ? (8 * g + 2 * v + lo)
                                   : (16 + 8 * g + 2 * (v - 4) + lo));
    float val = (k < K && n < Nw) ? W[(size_t)k * Nw + n] : 0.0f;
    P[idx] = (_Float16)val;
}

// ---------------------------------------------------------------------------
// Generic WMMA GEMM:  C[M x Nw] = epi(A[M x lda] @ Wpacked + bias)
//   * wave tile 16 x 64 (4 n-fragments), block = 8 waves stacked in M
//   * B panel (4 n-frags x kt, <= 60KB) staged once into LDS, read as v16h
//   * AF16 template: A either f32 (converted in-register) or f16 (direct)
//   * epilogue: optional bias, SiLU, per-row scale, multiply-into, f16 store
// Requirements: M % 16 == 0, lda >= Kp, Kp % 32 == 0 (pad cols of A zeroed).
// ---------------------------------------------------------------------------
template <int AF16>
__global__ __launch_bounds__(256)
void wmma_gemm_kernel(const void* __restrict__ Av, const _Float16* __restrict__ P,
                      const float* __restrict__ bias, void* __restrict__ Cv,
                      const float* __restrict__ row_scale,
                      int M, int Nw, int lda, int ldc, int Kp,
                      int out_f16, int do_silu, int mul_into) {
    __shared__ _Float16 Bs[15 * 2048];          // up to 60 KB B panel
    const int lane = threadIdx.x & 31;
    const int wave = threadIdx.x >> 5;
    const int kt   = Kp >> 5;
    const int n0   = blockIdx.y * 64;
    const int nt0  = n0 >> 4;

    // cooperative linear stage of the contiguous panel [nt0*kt, (nt0+4)*kt)
    {
        const v16h* srcp = (const v16h*)(P + (size_t)nt0 * kt * 512);
        v16h* dstp = (v16h*)Bs;
        const int chunks = kt * 128;            // 32B chunks in panel
        for (int c = threadIdx.x; c < chunks; c += 256) dstp[c] = srcp[c];
    }
    __syncthreads();

    const int m0 = blockIdx.x * 128 + wave * 16;
    if (m0 >= M) return;                        // wave-uniform: EXEC stays full

    const int g = lane >> 4;
    const size_t rowoff = (size_t)(m0 + (lane & 15)) * lda + 8 * g;
    const float*    Af = (const float*)Av;
    const _Float16* Ah = (const _Float16*)Av;

    v8f acc0 = {}, acc1 = {}, acc2 = {}, acc3 = {};
    for (int k = 0; k < kt; ++k) {
        v16h a;
        if (AF16) {
            const _Float16* ap = Ah + rowoff + k * 32;
            v8h lo = *(const v8h*)(ap);
            v8h hi = *(const v8h*)(ap + 16);
            a = __builtin_shufflevector(lo, hi, 0, 1, 2, 3, 4, 5, 6, 7,
                                        8, 9, 10, 11, 12, 13, 14, 15);
        } else {
            const float* ap = Af + rowoff + k * 32;
            float4 x0 = *(const float4*)(ap);
            float4 x1 = *(const float4*)(ap + 4);
            float4 x2 = *(const float4*)(ap + 16);
            float4 x3 = *(const float4*)(ap + 20);
            a[0]  = (_Float16)x0.x; a[1]  = (_Float16)x0.y;
            a[2]  = (_Float16)x0.z; a[3]  = (_Float16)x0.w;
            a[4]  = (_Float16)x1.x; a[5]  = (_Float16)x1.y;
            a[6]  = (_Float16)x1.z; a[7]  = (_Float16)x1.w;
            a[8]  = (_Float16)x2.x; a[9]  = (_Float16)x2.y;
            a[10] = (_Float16)x2.z; a[11] = (_Float16)x2.w;
            a[12] = (_Float16)x3.x; a[13] = (_Float16)x3.y;
            a[14] = (_Float16)x3.z; a[15] = (_Float16)x3.w;
        }
        v16h b0 = *(const v16h*)(Bs + (((0 * kt + k) * 32 + lane) * 16));
        v16h b1 = *(const v16h*)(Bs + (((1 * kt + k) * 32 + lane) * 16));
        v16h b2 = *(const v16h*)(Bs + (((2 * kt + k) * 32 + lane) * 16));
        v16h b3 = *(const v16h*)(Bs + (((3 * kt + k) * 32 + lane) * 16));

        acc0 = __builtin_amdgcn_wmma_f32_16x16x32_f16(false, a, false, b0, (short)0, acc0, false, false);
        acc1 = __builtin_amdgcn_wmma_f32_16x16x32_f16(false, a, false, b1, (short)0, acc1, false, false);
        acc2 = __builtin_amdgcn_wmma_f32_16x16x32_f16(false, a, false, b2, (short)0, acc2, false, false);
        acc3 = __builtin_amdgcn_wmma_f32_16x16x32_f16(false, a, false, b3, (short)0, acc3, false, false);
    }

    float*    Cf = (float*)Cv;
    _Float16* Ch = (_Float16*)Cv;
    v8f accs[4] = {acc0, acc1, acc2, acc3};
#pragma unroll
    for (int j = 0; j < 4; ++j) {
        int col = n0 + j * 16 + (lane & 15);
        if (col >= Nw) continue;                // Nw % 16 == 0 -> frag-uniform
        float bv = bias ? bias[col] : 0.0f;
#pragma unroll
        for (int r = 0; r < 8; ++r) {
            int orow = m0 + 8 * g + r;
            float v = accs[j][r] + bv;
            if (do_silu) v = silu_f(v);
            if (row_scale) v *= row_scale[orow];
            size_t idx = (size_t)orow * ldc + col;
            if (mul_into) v *= Cf[idx];
            if (out_f16) Ch[idx] = (_Float16)v;
            else         Cf[idx] = v;
        }
    }
}

// ---------------------------------------------------------------------------
// Per-edge geometry + RBF expansion (stored f16 for GEMM A) + local frame.
// ---------------------------------------------------------------------------
__global__ void edge_pre_kernel(const float* __restrict__ pos,
                                const int* __restrict__ src_i, const int* __restrict__ dst_i,
                                _Float16* __restrict__ rbf16, float* __restrict__ rb,
                                float* __restrict__ diff, float* __restrict__ frame) {
    int e = blockIdx.x * blockDim.x + threadIdx.x;
    if (e >= NEDGES) return;
    int s = src_i[e], d = dst_i[e];
    float sx = pos[s * 3], sy = pos[s * 3 + 1], sz = pos[s * 3 + 2];
    float dx = pos[d * 3], dy = pos[d * 3 + 1], dz = pos[d * 3 + 2];
    float rx = sx - dx, ry = sy - dy, rz = sz - dz;
    float dd = rx * rx + ry * ry + rz * rz + 1e-24f;
    float dist = sqrtf(dd);
    float rbv = 0.5f * (__cosf(dist * (PI_F / 5.0f)) + 1.0f);
    rb[e] = rbv;
    float rbounds = (dist < 5.0f) ? rbv : 0.0f;

    const float start = 0.0067379469990854670966f;           // exp(-5)
    const float bden  = (2.0f / 96.0f) * (1.0f - start);
    const float beta  = 1.0f / (bden * bden);
    float ed = __expf(-dist);
    for (int i = 0; i < RBF_N; ++i) {
        float mean = start + (1.0f - start) * ((float)i / 95.0f);
        float t = ed - mean;
        rbf16[(size_t)e * RBF_N + i] = (_Float16)(rbounds * __expf(-beta * t * t));
    }
    float inv = 1.0f / dist;
    float ex = rx * inv, ey = ry * inv, ez = rz * inv;      // edge_diff
    float cx = sy * dz - sz * dy, cy = sz * dx - sx * dz, cz = sx * dy - sy * dx;
    float cinv = 1.0f / sqrtf(cx * cx + cy * cy + cz * cz + 1e-24f);
    cx *= cinv; cy *= cinv; cz *= cinv;                     // edge_cross
    float vx = ey * cz - ez * cy, vy = ez * cx - ex * cz, vz = ex * cy - ey * cx;
    diff[e * 3] = ex; diff[e * 3 + 1] = ey; diff[e * 3 + 2] = ez;
    float* f = frame + (size_t)e * 9;                       // frame[a][k]
    f[0] = ex; f[1] = cx; f[2] = vx;
    f[3] = ey; f[4] = cy; f[5] = vy;
    f[6] = ez; f[7] = cz; f[8] = vz;
}

__global__ void s_init_kernel(const int* __restrict__ z, const float* __restrict__ z_emb,
                              float* __restrict__ s) {
    int i = blockIdx.x * blockDim.x + threadIdx.x;
    if (i >= NNODES * HDIM) return;
    int n = i >> 7, h = i & 127;
    s[i] = z_emb[z[n] * HDIM + h];
}

__global__ void s_scatter_kernel(const int* __restrict__ src_i, const int* __restrict__ dst_i,
                                 const int* __restrict__ z, const float* __restrict__ n_emb,
                                 const float* __restrict__ rh, float* __restrict__ s) {
    long long i = (long long)blockIdx.x * blockDim.x + threadIdx.x;
    if (i >= (long long)NEDGES * HDIM) return;
    int e = (int)(i >> 7), h = (int)(i & 127);
    atomicAdd(&s[(size_t)dst_i[e] * HDIM + h],
              n_emb[z[src_i[e]] * HDIM + h] * rh[i]);
}

__global__ void sij_scatter_kernel(const int* __restrict__ src_i, const int* __restrict__ dst_i,
                                   const float* __restrict__ rh, const float* __restrict__ diff,
                                   const float* __restrict__ s1, float* __restrict__ Sij) {
    long long i = (long long)blockIdx.x * blockDim.x + threadIdx.x;
    if (i >= (long long)NEDGES * HDIM) return;
    int e = (int)(i >> 7), h = (int)(i & 127);
    float val = rh[i] * s1[(size_t)src_i[e] * HDIM + h];
    size_t base = (size_t)dst_i[e] * 3 * HDIM + h;
#pragma unroll
    for (int a = 0; a < 3; ++a)
        atomicAdd(&Sij[base + a * HDIM], val * diff[e * 3 + a]);
}

// scalarize(): frame contraction + tiny 3->32->1 MLP, writes ew_pre16[:, 0:256]
__global__ void scalarize_kernel(const int* __restrict__ src_i, const int* __restrict__ dst_i,
                                 const float* __restrict__ Sij, const float* __restrict__ frame,
                                 const float* __restrict__ rb,
                                 const float* __restrict__ l1W, const float* __restrict__ l1b,
                                 const float* __restrict__ l2W, const float* __restrict__ l2b,
                                 _Float16* __restrict__ ew_pre16) {
    long long i = (long long)blockIdx.x * blockDim.x + threadIdx.x;
    if (i >= (long long)NEDGES * HDIM) return;
    int e = (int)(i >> 7), h = (int)(i & 127);
    const float* f = frame + (size_t)e * 9;
    float rbv = rb[e];
#pragma unroll
    for (int which = 0; which < 2; ++which) {
        int node = (which == 0) ? src_i[e] : dst_i[e];
        const float* Se = Sij + (size_t)node * 3 * HDIM;
        float a0 = Se[h], a1 = Se[HDIM + h], a2 = Se[2 * HDIM + h];
        float sc0 = a0 * f[0] + a1 * f[3] + a2 * f[6];
        float sc1 = fabsf(a0 * f[1] + a1 * f[4] + a2 * f[7]);
        float sc2 = a0 * f[2] + a1 * f[5] + a2 * f[8];
        float out = l2b[0];
        for (int j = 0; j < 32; ++j) {
            float hv = silu_f(sc0 * l1W[j] + sc1 * l1W[32 + j] + sc2 * l1W[64 + j] + l1b[j]);
            out += hv * l2W[j];
        }
        ew_pre16[(size_t)e * DIN + which * HDIM + h] = (_Float16)((out + sc0) * rbv);
    }
}

// ew_pre16[:, 256:384] = radial_hidden ; ew_pre16[:, 384:480] = rbf
__global__ void ew_fill_kernel(const float* __restrict__ rh, const _Float16* __restrict__ rbf16,
                               _Float16* __restrict__ ew_pre16) {
    long long i = (long long)blockIdx.x * blockDim.x + threadIdx.x;
    if (i >= (long long)NEDGES * (HDIM + RBF_N)) return;
    int e = (int)(i / (HDIM + RBF_N)), j = (int)(i % (HDIM + RBF_N));
    _Float16 v = (j < HDIM) ? (_Float16)rh[(size_t)e * HDIM + j]
                            : rbf16[(size_t)e * RBF_N + (j - HDIM)];
    ew_pre16[(size_t)e * DIN + 2 * HDIM + j] = v;
}

__global__ void message_kernel(const int* __restrict__ src_i, const int* __restrict__ dst_i,
                               const float* __restrict__ xh, const float* __restrict__ rbfh,
                               const float* __restrict__ vec, const float* __restrict__ diff,
                               float* __restrict__ s_acc, float* __restrict__ vec_delta) {
    long long i = (long long)blockIdx.x * blockDim.x + threadIdx.x;
    if (i >= (long long)NEDGES * HDIM) return;
    int e = (int)(i >> 7), h = (int)(i & 127);
    int se = src_i[e], de = dst_i[e];
    const float* xr = xh + (size_t)se * 3 * HDIM;
    const float* rr = rbfh + (size_t)e * 3 * HDIM;
    float x1  = xr[h] * rr[h];
    float xh2 = xr[HDIM + h] * rr[HDIM + h] * INV_SQRT3;
    float xh3 = xr[2 * HDIM + h] * rr[2 * HDIM + h];
    atomicAdd(&s_acc[(size_t)de * HDIM + h], x1);
    const float* vs = vec + (size_t)se * 3 * HDIM;
    const float* df = diff + (size_t)e * 3;
#pragma unroll
    for (int a = 0; a < 3; ++a) {
        float vmsg = (vs[a * HDIM + h] * xh2 + xh3 * df[a]) * INV_SQRTH;
        atomicAdd(&vec_delta[(size_t)de * 3 * HDIM + a * HDIM + h], vmsg);
    }
}

__global__ void node_update_kernel(float* __restrict__ s, const float* __restrict__ s_acc,
                                   float* __restrict__ vec, const float* __restrict__ vec_delta) {
    int i = blockIdx.x * blockDim.x + threadIdx.x;
    if (i >= NNODES * HDIM) return;
    int n = i >> 7, h = i & 127;
    s[i] = (s[i] + s_acc[i]) * INV_SQRT2;
#pragma unroll
    for (int a = 0; a < 3; ++a) {
        size_t k = (size_t)n * 3 * HDIM + a * HDIM + h;
        vec[k] += vec_delta[k];
    }
}

// cat16[n, 0:128] = s ; cat16[n, 128:256] = v1 = ||vm[n,:,h]||
__global__ void v1cat_kernel(const float* __restrict__ vm, const float* __restrict__ s,
                             _Float16* __restrict__ cat16) {
    int i = blockIdx.x * blockDim.x + threadIdx.x;
    if (i >= NNODES * HDIM) return;
    int n = i >> 7, h = i & 127;
    const float* vr = vm + (size_t)n * 3 * HDIM + h;
    float a0 = vr[0], a1 = vr[HDIM], a2 = vr[2 * HDIM];
    cat16[(size_t)n * 256 + h] = (_Float16)s[i];
    cat16[(size_t)n * 256 + HDIM + h] = (_Float16)sqrtf(a0 * a0 + a1 * a1 + a2 * a2 + 1e-24f);
}

__global__ void v2_kernel(const float* __restrict__ vec, const float* __restrict__ vec2W,
                          float* __restrict__ v2) {
    int i = blockIdx.x * blockDim.x + threadIdx.x;
    if (i >= NNODES * 3) return;
    const float* vr = vec + (size_t)(i / 3) * 3 * HDIM + (i % 3) * HDIM;
    float acc = 0.0f;
    for (int h = 0; h < HDIM; ++h) acc += vr[h] * vec2W[h];
    v2[i] = acc;
}

__global__ void final_kernel(const float* __restrict__ u1, const float* __restrict__ up2W,
                             const float* __restrict__ up2b, const float* __restrict__ v2,
                             float* __restrict__ out) {
    int i = blockIdx.x * blockDim.x + threadIdx.x;
    if (i >= NNODES * 3) return;
    int n = i / 3;
    const float* ur = u1 + (size_t)n * HDIM;
    float acc = up2b[1];
    for (int h = 0; h < HDIM; ++h) acc += ur[h] * up2W[h * 2 + 1];  // channel 1 only
    out[i] = acc * v2[i];
}

// ---------------------------------------------------------------------------
// Host orchestration
// ---------------------------------------------------------------------------
static inline int align_up_i(int x, int a) { return (x + a - 1) / a * a; }

extern "C" void kernel_launch(void* const* d_in, const int* in_sizes, int n_in,
                              void* d_out, int out_size, void* d_ws, size_t ws_size,
                              hipStream_t stream) {
    (void)in_sizes; (void)n_in; (void)out_size; (void)ws_size;

    // --- inputs, assuming JAX pytree flatten order (dict keys sorted) ------
    const int*   z     = (const int*)d_in[0];
    const float* pos   = (const float*)d_in[1];
    const int*   ei    = (const int*)d_in[2];
    const int*   srcI  = ei;             // edge_index[0]
    const int*   dstI  = ei + NEDGES;    // edge_index[1]
    const float* downW = (const float*)d_in[3];
    const float* downb = (const float*)d_in[4];
    const float *dirW[4], *dirb[4], *rbfW[4], *rbfb[4], *xp1W[4], *xp1b[4], *xp2W[4], *xp2b[4];
    for (int l = 0; l < 4; ++l) {
        int b = 5 + l * 8;
        dirW[l] = (const float*)d_in[b + 0]; dirb[l] = (const float*)d_in[b + 1];
        rbfW[l] = (const float*)d_in[b + 2]; rbfb[l] = (const float*)d_in[b + 3];
        xp1W[l] = (const float*)d_in[b + 4]; xp1b[l] = (const float*)d_in[b + 5];
        xp2W[l] = (const float*)d_in[b + 6]; xp2b[l] = (const float*)d_in[b + 7];
    }
    const float* lin1W = (const float*)d_in[37]; const float* lin1b = (const float*)d_in[38];
    const float* lin2W = (const float*)d_in[39]; const float* lin2b = (const float*)d_in[40];
    const float* n_emb = (const float*)d_in[41];
    const float* rad1W = (const float*)d_in[42]; const float* rad1b = (const float*)d_in[43];
    const float* rad2W = (const float*)d_in[44]; const float* rad2b = (const float*)d_in[45];
    const float* svW   = (const float*)d_in[46]; const float* svb   = (const float*)d_in[47];
    const float* up1W  = (const float*)d_in[48]; const float* up1b  = (const float*)d_in[49];
    const float* up2W  = (const float*)d_in[50]; const float* up2b  = (const float*)d_in[51];
    const float* vec1W = (const float*)d_in[52];
    const float* vec2W = (const float*)d_in[53];
    const float* z_emb = (const float*)d_in[54];

    // --- bump allocator over workspace ------------------------------------
    char* wsp = (char*)d_ws;
    size_t used = 0;
    auto alloc = [&](size_t bytes) -> void* {
        used = (used + 255) & ~(size_t)255;
        void* p = wsp + used;
        used += bytes;
        return p;
    };
    auto fbuf = [&](size_t n) { return (float*)alloc(n * sizeof(float)); };
    auto hbuf = [&](size_t n) { return (_Float16*)alloc(n * sizeof(_Float16)); };

    _Float16* rbf16  = hbuf((size_t)NEDGES * RBF_N);
    float* rb        = fbuf(NEDGES);
    float* diff      = fbuf((size_t)NEDGES * 3);
    float* frame     = fbuf((size_t)NEDGES * 9);
    _Float16* hid116 = hbuf((size_t)NEDGES * HDIM);
    float* rh        = fbuf((size_t)NEDGES * HDIM);      // radial_hidden
    float* s         = fbuf((size_t)NNODES * HDIM);
    float* s1        = fbuf((size_t)NNODES * HDIM);
    _Float16* t116   = hbuf((size_t)NNODES * HDIM);
    float* Sij       = fbuf((size_t)NNODES * 3 * HDIM);
    _Float16* ewp16  = hbuf((size_t)NEDGES * DIN);
    _Float16* ew16   = hbuf((size_t)NEDGES * 256);       // 240 cols + zero pad
    float* rbfh      = fbuf((size_t)NEDGES * 3 * HDIM);
    float* xh        = fbuf((size_t)NNODES * 3 * HDIM);
    float* s_acc     = fbuf((size_t)NNODES * HDIM);
    float* vec       = fbuf((size_t)NNODES * 3 * HDIM);
    float* vec_delta = fbuf((size_t)NNODES * 3 * HDIM);
    float* vm        = fbuf((size_t)NNODES * 3 * HDIM);
    _Float16* cat16  = hbuf((size_t)NNODES * 256);
    float* u1        = fbuf((size_t)NNODES * HDIM);
    float* v2        = fbuf((size_t)NNODES * 3);

    // --- pack all weights to WMMA B-fragment f16 layout -------------------
    auto pack = [&](const float* W, int K, int Nw) -> const _Float16* {
        int Kp = align_up_i(K, 32), Np = align_up_i(Nw, 64);
        size_t nel = (size_t)(Kp / 32) * (Np / 16) * 512;
        _Float16* P = (_Float16*)alloc(nel * sizeof(_Float16));
        pack_w_kernel<<<(unsigned)((nel + 255) / 256), 256, 0, stream>>>(W, P, K, Nw, Kp, Np);
        return P;
    };
    const _Float16* P_rad1 = pack(rad1W, RBF_N, HDIM);
    const _Float16* P_rad2 = pack(rad2W, HDIM, HDIM);
    const _Float16* P_sv   = pack(svW, HDIM, HDIM);
    const _Float16* P_down = pack(downW, DIN, DHALF);
    const _Float16* P_vec1 = pack(vec1W, HDIM, HDIM);
    const _Float16* P_up1  = pack(up1W, 2 * HDIM, HDIM);
    const _Float16 *P_xp1[4], *P_xp2[4], *P_rbf[4], *P_dir[4];
    for (int l = 0; l < 4; ++l) {
        P_xp1[l] = pack(xp1W[l], HDIM, HDIM);
        P_xp2[l] = pack(xp2W[l], HDIM, 3 * HDIM);
        P_rbf[l] = pack(rbfW[l], RBF_N, 3 * HDIM);
        P_dir[l] = pack(dirW[l], DHALF, 3 * HDIM);       // Kp=256, rows 240..255 zero
    }

    auto gemm = [&](const void* A, const _Float16* P, const float* bias, void* C,
                    const float* rscale, int M, int K, int Nw, int lda, int ldc,
                    int a_f16, int out_f16, int do_silu, int mul_into) {
        int Kp = align_up_i(K, 32), Np = align_up_i(Nw, 64);
        dim3 grid((M + 127) / 128, Np / 64);
        if (a_f16)
            wmma_gemm_kernel<1><<<grid, 256, 0, stream>>>(A, P, bias, C, rscale, M, Nw, lda, ldc,
                                                          Kp, out_f16, do_silu, mul_into);
        else
            wmma_gemm_kernel<0><<<grid, 256, 0, stream>>>(A, P, bias, C, rscale, M, Nw, lda, ldc,
                                                          Kp, out_f16, do_silu, mul_into);
    };
    auto egrid = [](long long n) { return (unsigned)((n + 255) / 256); };

    // --- pipeline ----------------------------------------------------------
    edge_pre_kernel<<<egrid(NEDGES), 256, 0, stream>>>(pos, srcI, dstI, rbf16, rb, diff, frame);

    // radial_hidden = rb * (silu(rbf@rad1+b) @ rad2 + b)
    gemm(rbf16, P_rad1, rad1b, hid116, nullptr, NEDGES, RBF_N, HDIM, RBF_N, HDIM, 1, 1, 1, 0);
    gemm(hid116, P_rad2, rad2b, rh, rb, NEDGES, HDIM, HDIM, HDIM, HDIM, 1, 0, 0, 0);

    // s = z_emb[z] + seg(n_emb[z[src]] * radial_hidden)
    s_init_kernel<<<egrid((long long)NNODES * HDIM), 256, 0, stream>>>(z, z_emb, s);
    s_scatter_kernel<<<egrid((long long)NEDGES * HDIM), 256, 0, stream>>>(srcI, dstI, z, n_emb, rh, s);

    // s1 = silu(s @ sv)
    gemm(s, P_sv, svb, s1, nullptr, NNODES, HDIM, HDIM, HDIM, HDIM, 0, 0, 1, 0);

    // Sij = seg(radial_hidden ⊗ edge_diff * s1[src])
    hipMemsetAsync(Sij, 0, (size_t)NNODES * 3 * HDIM * sizeof(float), stream);
    sij_scatter_kernel<<<egrid((long long)NEDGES * HDIM), 256, 0, stream>>>(srcI, dstI, rh, diff, s1, Sij);

    // ew_pre = [scalar3, scalar4, radial_hidden, rbf]  (f16)
    scalarize_kernel<<<egrid((long long)NEDGES * HDIM), 256, 0, stream>>>(
        srcI, dstI, Sij, frame, rb, lin1W, lin1b, lin2W, lin2b, ewp16);
    ew_fill_kernel<<<egrid((long long)NEDGES * (HDIM + RBF_N)), 256, 0, stream>>>(rh, rbf16, ewp16);

    // ew = silu(ew_pre @ down)  (f16, 240 cols in ld 256, pad zeroed)
    hipMemsetAsync(ew16, 0, (size_t)NEDGES * 256 * sizeof(_Float16), stream);
    gemm(ewp16, P_down, downb, ew16, nullptr, NEDGES, DIN, DHALF, DIN, 256, 1, 1, 1, 0);

    hipMemsetAsync(vec, 0, (size_t)NNODES * 3 * HDIM * sizeof(float), stream);

    for (int l = 0; l < NLAYERS; ++l) {
        gemm(s, P_xp1[l], xp1b[l], t116, nullptr, NNODES, HDIM, HDIM, HDIM, HDIM, 0, 1, 1, 0);
        gemm(t116, P_xp2[l], xp2b[l], xh, nullptr, NNODES, HDIM, 3 * HDIM, HDIM, 3 * HDIM, 1, 0, 0, 0);
        // rbfh = (rbf @ rbfW + b) * (ew @ dirW + b)
        gemm(rbf16, P_rbf[l], rbfb[l], rbfh, nullptr, NEDGES, RBF_N, 3 * HDIM, RBF_N, 3 * HDIM, 1, 0, 0, 0);
        gemm(ew16, P_dir[l], dirb[l], rbfh, nullptr, NEDGES, DHALF, 3 * HDIM, 256, 3 * HDIM, 1, 0, 0, 1);

        hipMemsetAsync(s_acc, 0, (size_t)NNODES * HDIM * sizeof(float), stream);
        hipMemsetAsync(vec_delta, 0, (size_t)NNODES * 3 * HDIM * sizeof(float), stream);
        message_kernel<<<egrid((long long)NEDGES * HDIM), 256, 0, stream>>>(
            srcI, dstI, xh, rbfh, vec, diff, s_acc, vec_delta);
        node_update_kernel<<<egrid((long long)NNODES * HDIM), 256, 0, stream>>>(s, s_acc, vec, vec_delta);
    }

    // heads
    gemm(vec, P_vec1, nullptr, vm, nullptr, NNODES * 3, HDIM, HDIM, HDIM, HDIM, 0, 0, 0, 0);
    v1cat_kernel<<<egrid((long long)NNODES * HDIM), 256, 0, stream>>>(vm, s, cat16);
    gemm(cat16, P_up1, up1b, u1, nullptr, NNODES, 2 * HDIM, HDIM, 2 * HDIM, HDIM, 1, 0, 1, 0);
    v2_kernel<<<egrid((long long)NNODES * 3), 256, 0, stream>>>(vec, vec2W, v2);
    final_kernel<<<egrid((long long)NNODES * 3), 256, 0, stream>>>(u1, up2W, up2b, v2, (float*)d_out);
}
